// GVAE_60266981097985
// MI455X (gfx1250) — matrix operand order repaired and checked by reading, not compiled, not run on hardware
//
#include <hip/hip_runtime.h>
#include <hip/hip_bf16.h>

// ---------------------------------------------------------------------------
// GVAE (GATv2 x2 + mu/lv heads + z z^T decode) for MI455X (gfx1250, wave32).
// Matrix work uses v_wmma_f32_16x16x32_bf16 with ISA-table VGPR layouts.
// Each wave computes a 16x(16*NT) output strip to reuse the A fragment.
// ---------------------------------------------------------------------------

typedef __attribute__((ext_vector_type(8)))  __bf16 v8bf;
typedef __attribute__((ext_vector_type(16))) __bf16 v16bf;
typedef __attribute__((ext_vector_type(8)))  float  v8f;

#define GN   10000      // nodes
#define GE   320000     // edges (before self loops)
#define GEL  (GE + GN)  // edges incl. self loops
#define DIN  256
#define HID  64
#define HEADS 4
#define HC   (HEADS * HID)  // 256
#define LAT  32
#define NEG_SLOPE 0.2f
#define NT   4          // N-tiles per wave (A-fragment reuse factor)

// ---------------- order-preserving float <-> uint for atomic max ------------
__device__ __forceinline__ unsigned ordf(float f) {
  unsigned u = __float_as_uint(f);
  return (u & 0x80000000u) ? ~u : (u | 0x80000000u);
}
__device__ __forceinline__ float unordf(unsigned u) {
  return (u & 0x80000000u) ? __uint_as_float(u & 0x7fffffffu) : __uint_as_float(~u);
}

// ---------------- WMMA GEMM: C[M,Ncols] = A[M,K](bf16) @ Wt[Ncols,K]^T + b --
// Wt is the B operand stored column-major (Wt[n*K + k] = B[k][n]), so both A
// and B fragments are contiguous 16-byte v8bf loads per lane.
// A frag (16x32):  lane r=lane&15 -> row, g=lane>>4; k = g*8 + i (i<8),
//                  k = 16 + g*8 + (i-8) (i>=8)           [ISA 7.12.2]
// B frag (32x16):  lane -> col, k same as A frag (g*8 + i / 16 + g*8 + (i-8))
// C/D f32 (16x16): vgpr rr -> row g*8+rr, lane -> col     [ISA 7.12.2]
__global__ void wmma_gemm_bf16(const __bf16* __restrict__ A,
                               const __bf16* __restrict__ Wt,
                               const float* __restrict__ bias,  // [Ncols] or null
                               float* __restrict__ C,
                               int M, int K, int Ncols)
{
  const int nTilesN = Ncols >> 4;
  const int nStrips = (nTilesN + NT - 1) / NT;
  const int wave = blockIdx.x * (blockDim.x >> 5) + (threadIdx.x >> 5);
  const int nWaves = (M >> 4) * nStrips;
  if (wave >= nWaves) return;          // wave-uniform: EXEC stays all-ones
  const int lane = threadIdx.x & 31;
  const int g = lane >> 4;
  const int r = lane & 15;
  const int tm = wave / nStrips;
  const int tnBase = (wave - tm * nStrips) * NT;

  const __bf16* __restrict__ arow = A + (size_t)(tm * 16 + r) * K;
  const __bf16* bcol[NT];
#pragma unroll
  for (int i = 0; i < NT; ++i) {
    int tn = tnBase + i;
    if (tn >= nTilesN) tn = nTilesN - 1;     // clamp: wave-uniform, loads stay in range
    bcol[i] = Wt + (size_t)(tn * 16 + r) * K;
  }

  v8f c[NT] = {};
  for (int kb = 0; kb < K; kb += 32) {
    if (kb + 32 < K) {
      __builtin_prefetch(arow + kb + 32, 0, 3);           // WGP-scope prefetch
      __builtin_prefetch(bcol[0] + kb + 32, 0, 3);
      __builtin_prefetch(bcol[NT - 1] + kb + 32, 0, 3);
    }
    v8bf a0 = *(const v8bf*)(arow + kb + g * 8);
    v8bf a1 = *(const v8bf*)(arow + kb + 16 + g * 8);
    v16bf a = __builtin_shufflevector(a0, a1, 0,1,2,3,4,5,6,7,8,9,10,11,12,13,14,15);
#pragma unroll
    for (int i = 0; i < NT; ++i) {
      v8bf b0 = *(const v8bf*)(bcol[i] + kb + g * 8);
      v8bf b1 = *(const v8bf*)(bcol[i] + kb + 16 + g * 8);
      v16bf b = __builtin_shufflevector(b0, b1, 0,1,2,3,4,5,6,7,8,9,10,11,12,13,14,15);
      c[i] = __builtin_amdgcn_wmma_f32_16x16x32_bf16(
          /*neg_a=*/false, a, /*neg_b=*/false, b,
          /*c_mod=*/(short)0, c[i], /*reuse_a=*/false, /*reuse_b=*/false);
    }
  }

  const int rowBase = tm * 16 + g * 8;
#pragma unroll
  for (int i = 0; i < NT; ++i) {
    const int tn = tnBase + i;
    if (tn >= nTilesN) break;                // wave-uniform guard for last strip
    const int col = tn * 16 + r;
    const float bv = bias ? bias[col] : 0.0f;
    float* __restrict__ crow = C + (size_t)rowBase * Ncols + col;
#pragma unroll
    for (int rr = 0; rr < 8; ++rr)
      crow[(size_t)rr * Ncols] = c[i][rr] + bv;
  }
}

// ---------------- small utility kernels -------------------------------------
__global__ void k_fill_f32(float* p, float v, int n) {
  int i = blockIdx.x * blockDim.x + threadIdx.x;
  if (i < n) p[i] = v;
}
__global__ void k_fill_u32(unsigned* p, unsigned v, int n) {
  int i = blockIdx.x * blockDim.x + threadIdx.x;
  if (i < n) p[i] = v;
}
__global__ void k_f32_to_bf16(const float* __restrict__ in, __bf16* __restrict__ out, int n) {
  int i = blockIdx.x * blockDim.x + threadIdx.x;
  if (i < n) out[i] = (__bf16)in[i];
}
// Wt[n*K + k] = bf16(W[k*Ncols + n])
__global__ void k_transpose_bf16(const float* __restrict__ W, __bf16* __restrict__ Wt,
                                 int K, int Ncols) {
  int i = blockIdx.x * blockDim.x + threadIdx.x;
  if (i >= K * Ncols) return;
  int k = i / Ncols, n = i - k * Ncols;
  Wt[(size_t)n * K + k] = (__bf16)W[(size_t)k * Ncols + n];
}

// ---------------- GATv2 attention (3 passes over edges) ---------------------
// pass 1: score[e,h] = att_h . leakyrelu(xl[src] + xr[dst]); segment max
__global__ void k_edge_score(const int* __restrict__ ei,
                             const float* __restrict__ xl, const float* __restrict__ xr,
                             const float* __restrict__ att,
                             float* __restrict__ score, unsigned* __restrict__ smaxU,
                             int H, int Cc)
{
  int t = blockIdx.x * blockDim.x + threadIdx.x;
  if (t >= GEL * H) return;
  int e = t / H, h = t - e * H;
  int src = (e < GE) ? ei[e]      : (e - GE);
  int dst = (e < GE) ? ei[GE + e] : (e - GE);
  int hc = H * Cc;
  const float* pl = xl + (size_t)src * hc + h * Cc;
  const float* pr = xr + (size_t)dst * hc + h * Cc;
  const float* pa = att + h * Cc;
  float s = 0.f;
  for (int c = 0; c < Cc; ++c) {
    float v = pl[c] + pr[c];
    v = (v > 0.f) ? v : NEG_SLOPE * v;
    s = fmaf(pa[c], v, s);
  }
  score[t] = s;
  atomicMax(smaxU + (size_t)dst * H + h, ordf(s));
}

// pass 2: ex = exp(score - smax[dst]); segment sum -> denom
__global__ void k_edge_softmax(const int* __restrict__ ei,
                               float* __restrict__ score,
                               const unsigned* __restrict__ smaxU,
                               float* __restrict__ denom, int H)
{
  int t = blockIdx.x * blockDim.x + threadIdx.x;
  if (t >= GEL * H) return;
  int e = t / H, h = t - e * H;
  int dst = (e < GE) ? ei[GE + e] : (e - GE);
  float m = unordf(smaxU[(size_t)dst * H + h]);
  float ex = expf(score[t] - m);
  score[t] = ex;
  atomicAdd(denom + (size_t)dst * H + h, ex);
}

// pass 3: out[dst,h,c] += (ex/denom) * xl[src,h,c]
__global__ void k_edge_aggregate(const int* __restrict__ ei,
                                 const float* __restrict__ score,
                                 const float* __restrict__ denom,
                                 const float* __restrict__ xl,
                                 float* __restrict__ out, int H, int Cc)
{
  int hc = H * Cc;
  int t = blockIdx.x * blockDim.x + threadIdx.x;
  if (t >= GEL * hc) return;
  int e = t / hc, j = t - e * hc;
  int h = j / Cc;
  int src = (e < GE) ? ei[e]      : (e - GE);
  int dst = (e < GE) ? ei[GE + e] : (e - GE);
  float alpha = score[(size_t)e * H + h] / (denom[(size_t)dst * H + h] + 1e-16f);
  atomicAdd(out + (size_t)dst * hc + j, alpha * xl[(size_t)src * hc + j]);
}

// bias (+ optional ELU), in place
__global__ void k_bias_act(float* __restrict__ out, const float* __restrict__ bo,
                           int n, int ncols, int do_elu)
{
  int i = blockIdx.x * blockDim.x + threadIdx.x;
  if (i >= n) return;
  float v = out[i] + bo[i % ncols];
  if (do_elu) v = (v > 0.f) ? v : expm1f(v);
  out[i] = v;
}

// z = mu + eps * exp(0.5*lv), stored as bf16 for the WMMA decode
__global__ void k_reparam_bf16(const float* __restrict__ mu, const float* __restrict__ lv,
                               const float* __restrict__ eps, __bf16* __restrict__ zb, int n)
{
  int i = blockIdx.x * blockDim.x + threadIdx.x;
  if (i < n) zb[i] = (__bf16)(mu[i] + eps[i] * expf(0.5f * lv[i]));
}

// ---------------- workspace layout (all sizes multiples of 256 B) -----------
static constexpr size_t OFF_HB    = 0;                                  // bf16 [N,256]
static constexpr size_t OFF_WLT   = OFF_HB    + (size_t)GN * 256 * 2;   // bf16 [256,256]
static constexpr size_t OFF_WRT   = OFF_WLT   + 256 * 256 * 2;
static constexpr size_t OFF_XL    = OFF_WRT   + 256 * 256 * 2;          // f32 [N,256]
static constexpr size_t OFF_XR    = OFF_XL    + (size_t)GN * 256 * 4;
static constexpr size_t OFF_SCORE = OFF_XR    + (size_t)GN * 256 * 4;   // f32 [GEL,4]
static constexpr size_t OFF_SMAX  = OFF_SCORE + (size_t)GEL * HEADS * 4;
static constexpr size_t OFF_DENOM = OFF_SMAX  + (size_t)GN * HEADS * 4;
static constexpr size_t OFF_HOUT  = OFF_DENOM + (size_t)GN * HEADS * 4; // f32 [N,256]
static constexpr size_t OFF_ZB    = OFF_HOUT  + (size_t)GN * 256 * 4;   // bf16 [N,32]
// total ~42.3 MB

static inline int cdiv(long long a, long long b) { return (int)((a + b - 1) / b); }

static void run_gemm(const __bf16* A, const __bf16* Wt, const float* bias, float* C,
                     int M, int K, int Ncols, hipStream_t s)
{
  int nStrips = cdiv(Ncols / 16, NT);
  int nWaves  = (M / 16) * nStrips;
  int blocks  = cdiv(nWaves, 4);                  // 128 threads = 4 waves/block
  wmma_gemm_bf16<<<blocks, 128, 0, s>>>(A, Wt, bias, C, M, K, Ncols);
}

static void run_gat_layer(const int* ei, const __bf16* hb,
                          const float* Wl, const float* bl,
                          const float* Wr, const float* br,
                          const float* att, const float* bo,
                          int H, int Cc, int do_elu,
                          float* out,                    // [N, H*Cc]
                          char* ws, hipStream_t s)
{
  const int hc = H * Cc;
  __bf16* WlT = (__bf16*)(ws + OFF_WLT);
  __bf16* WrT = (__bf16*)(ws + OFF_WRT);
  float*  xl  = (float*)(ws + OFF_XL);
  float*  xr  = (float*)(ws + OFF_XR);
  float*  sc  = (float*)(ws + OFF_SCORE);
  unsigned* sm = (unsigned*)(ws + OFF_SMAX);
  float*  dn  = (float*)(ws + OFF_DENOM);

  // weights -> transposed bf16
  k_transpose_bf16<<<cdiv(256 * hc, 256), 256, 0, s>>>(Wl, WlT, 256, hc);
  k_transpose_bf16<<<cdiv(256 * hc, 256), 256, 0, s>>>(Wr, WrT, 256, hc);
  // GEMMs: xl = h @ Wl + bl, xr = h @ Wr + br
  run_gemm(hb, WlT, bl, xl, GN, 256, hc, s);
  run_gemm(hb, WrT, br, xr, GN, 256, hc, s);
  // init segment state + output accumulator
  k_fill_u32<<<cdiv(GN * H, 256), 256, 0, s>>>(sm, 0u, GN * H);
  k_fill_f32<<<cdiv(GN * H, 256), 256, 0, s>>>(dn, 0.f, GN * H);
  k_fill_f32<<<cdiv((long long)GN * hc, 256), 256, 0, s>>>(out, 0.f, GN * hc);
  // 3-pass edge softmax attention
  k_edge_score    <<<cdiv((long long)GEL * H, 256), 256, 0, s>>>(ei, xl, xr, att, sc, sm, H, Cc);
  k_edge_softmax  <<<cdiv((long long)GEL * H, 256), 256, 0, s>>>(ei, sc, sm, dn, H);
  k_edge_aggregate<<<cdiv((long long)GEL * hc, 256), 256, 0, s>>>(ei, sc, dn, xl, out, H, Cc);
  k_bias_act      <<<cdiv((long long)GN * hc, 256), 256, 0, s>>>(out, bo, GN * hc, hc, do_elu);
}

extern "C" void kernel_launch(void* const* d_in, const int* in_sizes, int n_in,
                              void* d_out, int out_size, void* d_ws, size_t ws_size,
                              hipStream_t stream) {
  const float* x   = (const float*)d_in[0];
  const int*   ei  = (const int*)  d_in[1];
  const float* eps = (const float*)d_in[2];
  const float *Wl[4], *bl[4], *Wr[4], *br[4], *att[4], *bo[4];
  for (int L = 0; L < 4; ++L) {     // L: 0=l0, 1=l1, 2=mu, 3=lv
    Wl[L]  = (const float*)d_in[3 + 6 * L + 0];
    bl[L]  = (const float*)d_in[3 + 6 * L + 1];
    Wr[L]  = (const float*)d_in[3 + 6 * L + 2];
    br[L]  = (const float*)d_in[3 + 6 * L + 3];
    att[L] = (const float*)d_in[3 + 6 * L + 4];
    bo[L]  = (const float*)d_in[3 + 6 * L + 5];
  }
  char* ws = (char*)d_ws;
  __bf16* hb   = (__bf16*)(ws + OFF_HB);
  float*  hout = (float*)(ws + OFF_HOUT);
  __bf16* zb   = (__bf16*)(ws + OFF_ZB);

  float* adj = (float*)d_out;                       // [N,N]
  float* mu  = adj + (size_t)GN * GN;               // [N,LAT]
  float* lv  = mu + (size_t)GN * LAT;               // [N,LAT]

  // ---- layer 0: h1 = elu(gatv2(x)) -> hout
  k_f32_to_bf16<<<cdiv((long long)GN * DIN, 256), 256, 0, stream>>>(x, hb, GN * DIN);
  run_gat_layer(ei, hb, Wl[0], bl[0], Wr[0], br[0], att[0], bo[0],
                HEADS, HID, /*elu=*/1, hout, ws, stream);

  // ---- layer 1: h2 = elu(gatv2(h1)) -> hout (rotation via hb)
  k_f32_to_bf16<<<cdiv((long long)GN * HC, 256), 256, 0, stream>>>(hout, hb, GN * HC);
  run_gat_layer(ei, hb, Wl[1], bl[1], Wr[1], br[1], att[1], bo[1],
                HEADS, HID, /*elu=*/1, hout, ws, stream);

  // ---- mu / log_var heads (H=1, C=32, no activation), accumulate into d_out
  k_f32_to_bf16<<<cdiv((long long)GN * HC, 256), 256, 0, stream>>>(hout, hb, GN * HC);
  run_gat_layer(ei, hb, Wl[2], bl[2], Wr[2], br[2], att[2], bo[2],
                1, LAT, /*elu=*/0, mu, ws, stream);
  run_gat_layer(ei, hb, Wl[3], bl[3], Wr[3], br[3], att[3], bo[3],
                1, LAT, /*elu=*/0, lv, ws, stream);

  // ---- z = mu + eps * exp(0.5*lv), as bf16
  k_reparam_bf16<<<cdiv((long long)GN * LAT, 256), 256, 0, stream>>>(mu, lv, eps, zb, GN * LAT);

  // ---- adj = z @ z^T : zb serves as both A [N,32] and column-major B [N,32]
  run_gemm(zb, zb, /*bias=*/nullptr, adj, GN, LAT, GN, stream);
}
